// FusedGCNLayer_29222957482548
// MI455X (gfx1250) — matrix operand order repaired and checked by reading, not compiled, not run on hardware
//
#include <hip/hip_runtime.h>
#include <hip/hip_bf16.h>

typedef __attribute__((ext_vector_type(16))) __bf16 v16bf;
typedef __attribute__((ext_vector_type(8)))  __bf16 v8bf;
typedef __attribute__((ext_vector_type(8)))  float  v8f;

#define FEAT 256
#define EMB  256

// ---------------------------------------------------------------------------
// Kernel P: one-time split of W (fp32) into bf16 hi + residual lo.
// Removes all W conversion VALU from the 6250 GEMM workgroups.
// ---------------------------------------------------------------------------
__global__ void gcn_pack_w(const float* __restrict__ w,
                           __bf16* __restrict__ w_hi,
                           __bf16* __restrict__ w_lo, int n)
{
    int i = blockIdx.x * blockDim.x + threadIdx.x;
    if (i < n) {
        const float  f  = w[i];
        const __bf16 hi = (__bf16)f;
        w_hi[i] = hi;
        w_lo[i] = (__bf16)(f - (float)hi);
    }
}

// ---------------------------------------------------------------------------
// Kernel 1: h[n,e] = sum_f x[n,f] * w[e,f]   (x @ W^T), bf16x3 split WMMA.
// One 16-row M tile per workgroup; 4 waves, each owns a 64-column span.
// The x tile is converted ONCE into LDS (hi/lo bf16) by all 128 threads,
// then fragments come from ds_load_b128; B fragments are straight v16bf
// loads of the pre-packed W (L2-resident).  Inner loop = loads + 12 WMMA.
// ---------------------------------------------------------------------------
__global__ __launch_bounds__(128) void gcn_gemm_bf16x3(
    const float* __restrict__ x,
    const __bf16* __restrict__ w_hi, const __bf16* __restrict__ w_lo,
    float* __restrict__ h, int n_nodes)
{
    __shared__ __bf16 sA_hi[16 * FEAT];   // 8 KB
    __shared__ __bf16 sA_lo[16 * FEAT];   // 8 KB

    const int tid    = threadIdx.x;
    const int lane   = tid & 31;
    const int wave   = tid >> 5;
    const int m_base = blockIdx.x * 16;
    if (m_base >= n_nodes) return;
    const int n_wave = wave * 64;

    // ---- stage + split the 16x256 A tile into LDS (32 elems per thread) ----
    {
        const int base = tid * 32;                 // 4096 / 128
        const int row  = base >> 8;
        const int col  = base & 255;
        int grow = m_base + row;
        if (grow >= n_nodes) grow = n_nodes - 1;   // tail clamp (loads only)
        const float* px = x + (size_t)grow * FEAT + col;
        #pragma unroll
        for (int q = 0; q < 8; ++q) {
            const float4 f = *(const float4*)(px + q * 4);
            const float run[4] = {f.x, f.y, f.z, f.w};
            #pragma unroll
            for (int t = 0; t < 4; ++t) {
                const __bf16 hi = (__bf16)run[t];
                sA_hi[base + q * 4 + t] = hi;
                sA_lo[base + q * 4 + t] = (__bf16)(run[t] - (float)hi);
            }
        }
    }
    __syncthreads();

    // A-fragment (16x32 bf16): lane holds row (lane&15); two contiguous
    // 8-element K runs, K-half chosen by lane>>4.
    const int arow  = lane & 15;
    const int khalf = lane >> 4;
    // B-fragment (32x16 bf16): lane holds column (lane&15) of W^T = a
    // contiguous 16-bf16 run of row W[e]; K sub-base from lane>>4.
    const int ncol = lane & 15;
    const int kb2  = (lane >> 4) * 16;

    v8f acc[4] = {};                                // 4 N-tiles x 8 VGPRs f32

    for (int kt = 0; kt < FEAT; kt += 32) {
        v16bf a_hi, a_lo;
        {
            const __bf16* pa = &sA_hi[arow * FEAT + kt + khalf * 16];
            *(v8bf*)&a_hi       = *(const v8bf*)pa;         // K 0..7   (+16*half)
            *((v8bf*)&a_hi + 1) = *(const v8bf*)(pa + 8);   // K 8..15  (+16*half)
            const __bf16* pl = &sA_lo[arow * FEAT + kt + khalf * 16];
            *(v8bf*)&a_lo       = *(const v8bf*)pl;
            *((v8bf*)&a_lo + 1) = *(const v8bf*)(pl + 8);
        }

        #pragma unroll
        for (int nt = 0; nt < 4; ++nt) {
            const size_t off =
                (size_t)(n_wave + nt * 16 + ncol) * FEAT + kt + kb2;
            const v16bf b_hi = *(const v16bf*)(w_hi + off);
            const v16bf b_lo = *(const v16bf*)(w_lo + off);
            acc[nt] = __builtin_amdgcn_wmma_f32_16x16x32_bf16(
                false, a_hi, false, b_hi, (short)0, acc[nt], false, false);
            acc[nt] = __builtin_amdgcn_wmma_f32_16x16x32_bf16(
                false, a_hi, false, b_lo, (short)0, acc[nt], false, false);
            acc[nt] = __builtin_amdgcn_wmma_f32_16x16x32_bf16(
                false, a_lo, false, b_hi, (short)0, acc[nt], false, false);
        }
    }

    // ---- store D (32-bit C/D layout: VGPR r -> row r + (lane>>4)*8) --------
    const int srow = m_base + (lane >> 4) * 8;
    #pragma unroll
    for (int nt = 0; nt < 4; ++nt) {
        #pragma unroll
        for (int r = 0; r < 8; ++r) {
            const int row = srow + r;
            if (row < n_nodes)
                h[(size_t)row * EMB + n_wave + nt * 16 + ncol] = acc[nt][r];
        }
    }
}

// ---------------------------------------------------------------------------
// Kernel 2: out = segment_sum(h[src] * val, dst).  edge_dst is sorted ->
// register run-accumulation, hardware f32 atomic flush on dst change.
// fgrp = tid&63 -> float4 feature slice (512B coalesced gathers per wave);
// sub = tid>>6 (wave-uniform) -> 64-edge sub-range.
// ---------------------------------------------------------------------------
__global__ __launch_bounds__(256) void gcn_spmm_scatter(
    const float* __restrict__ h, const int* __restrict__ esrc,
    const int* __restrict__ edst, const float* __restrict__ eval,
    float* __restrict__ out, int n_edges)
{
    const int fgrp = threadIdx.x & 63;
    const int sub  = threadIdx.x >> 6;
    int e = blockIdx.x * 256 + sub * 64;
    int e_end = e + 64;
    if (e_end > n_edges) e_end = n_edges;

    float4 acc = make_float4(0.f, 0.f, 0.f, 0.f);
    int cur = -1;
    for (; e < e_end; ++e) {
        const int d = edst[e];
        if (d != cur) {
            if (cur >= 0) {
                float* o = out + (size_t)cur * EMB + fgrp * 4;
                unsafeAtomicAdd(o + 0, acc.x);
                unsafeAtomicAdd(o + 1, acc.y);
                unsafeAtomicAdd(o + 2, acc.z);
                unsafeAtomicAdd(o + 3, acc.w);
                acc = make_float4(0.f, 0.f, 0.f, 0.f);
            }
            cur = d;
        }
        const int    s  = esrc[e];
        const float  v  = eval[e];
        const float4 hv = *(const float4*)(h + (size_t)s * EMB + fgrp * 4);
        acc.x += v * hv.x;
        acc.y += v * hv.y;
        acc.z += v * hv.z;
        acc.w += v * hv.w;
    }
    if (cur >= 0) {
        float* o = out + (size_t)cur * EMB + fgrp * 4;
        unsafeAtomicAdd(o + 0, acc.x);
        unsafeAtomicAdd(o + 1, acc.y);
        unsafeAtomicAdd(o + 2, acc.z);
        unsafeAtomicAdd(o + 3, acc.w);
    }
}

// ---------------------------------------------------------------------------
__global__ void gcn_zero_f32(float* __restrict__ p, size_t n)
{
    size_t i      = (size_t)blockIdx.x * blockDim.x + threadIdx.x;
    size_t stride = (size_t)gridDim.x * blockDim.x;
    for (; i < n; i += stride) p[i] = 0.0f;
}

// ---------------------------------------------------------------------------
extern "C" void kernel_launch(void* const* d_in, const int* in_sizes, int n_in,
                              void* d_out, int out_size, void* d_ws, size_t ws_size,
                              hipStream_t stream)
{
    const float* x    = (const float*)d_in[0];
    const float* w    = (const float*)d_in[1];
    const int*   esrc = (const int*)d_in[2];
    const int*   edst = (const int*)d_in[3];
    const float* eval = (const float*)d_in[4];
    float*       out  = (float*)d_out;

    const int n_nodes = in_sizes[0] / FEAT;
    const int n_edges = in_sizes[2];
    const int n_w     = in_sizes[1];

    // workspace layout: h [n_nodes*EMB f32] | w_hi | w_lo [EMB*FEAT bf16 each]
    char*  base    = (char*)d_ws;
    float* h       = (float*)base;
    size_t h_bytes = ((size_t)n_nodes * EMB * sizeof(float) + 255) & ~(size_t)255;
    __bf16* w_hi   = (__bf16*)(base + h_bytes);
    __bf16* w_lo   = w_hi + (size_t)EMB * FEAT;

    // out is poisoned by the harness -> zero it (scatter uses atomics).
    gcn_zero_f32<<<2048, 256, 0, stream>>>(out, (size_t)out_size);

    gcn_pack_w<<<(n_w + 255) / 256, 256, 0, stream>>>(w, w_hi, w_lo, n_w);

    const int mtiles = (n_nodes + 15) / 16;
    gcn_gemm_bf16x3<<<mtiles, 128, 0, stream>>>(x, w_hi, w_lo, h, n_nodes);

    const int eblocks = (n_edges + 255) / 256;
    gcn_spmm_scatter<<<eblocks, 256, 0, stream>>>(h, esrc, edst, eval, out,
                                                  n_edges);
}
